// CausalSelfAttention_32890859553073
// MI455X (gfx1250) — matrix operand order repaired and checked by reading, not compiled, not run on hardware
//
#include <hip/hip_runtime.h>
#include <hip/hip_bf16.h>

// ---------------------------------------------------------------------------
// CausalSelfAttention forward for MI455X (gfx1250, wave32, WMMA).
// B=2, T=2048, C=2048, H=16, HD=128.
// Pipeline: f32->bf16 cvt -> QKV GEMM (WMMA bf16, double-buffered LDS) ->
// RoPE+head-reorg (V stored transposed [HD,T]) -> flash attention (WMMA bf16
// QK^T and P*V, online softmax, prefetched K/V fragments) -> output GEMM.
// Workspace use: ~151 MB (L2-resident).
// ---------------------------------------------------------------------------

typedef __attribute__((ext_vector_type(16))) __bf16 v16bf;
typedef __attribute__((ext_vector_type(8)))  __bf16 bf8v;
typedef __attribute__((ext_vector_type(8)))  float  v8f;

#define Bz  2
#define Tz  2048
#define Cz  2048
#define Hz  16
#define HDz 128
#define C3z 6144

union frag_u { v16bf v; bf8v h[2]; };

// A-matrix fragment (16x32 bf16, ISA 7.12.2): lane l, row m=l&15.
// elements 0..7  -> k = kb8 + e        (kb8 = 0 for lanes 0-15, 8 for 16-31)
// elements 8..15 -> k = 16 + kb8 + e-8
__device__ inline v16bf load_frag_a(const __bf16* base, int stride, int m, int kb8) {
  frag_u u;
  u.h[0] = *(const bf8v*)(base + m * stride + kb8);
  u.h[1] = *(const bf8v*)(base + m * stride + kb8 + 16);
  return u.v;
}

// B-matrix fragment (32x16 bf16): lane l, col n=l&15.
// elements 0..7 -> k = kb16/2 + e, 8..15 -> k = 16 + kb16/2 + e-8 (kb16 = 0 for lanes 0-15, 16 for 16-31)
// "base" is row-major [N rows][stride] holding B^T (i.e. W[n][k]).
__device__ inline v16bf load_frag_b(const __bf16* base, int stride, int n, int kb16) {
  frag_u u;
  u.h[0] = *(const bf8v*)(base + n * stride + (kb16 >> 1));
  u.h[1] = *(const bf8v*)(base + n * stride + (kb16 >> 1) + 16);
  return u.v;
}

__device__ inline v8f wmma_bf16(v16bf a, v16bf b, v8f c) {
  return __builtin_amdgcn_wmma_f32_16x16x32_bf16(false, a, false, b, (short)0, c,
                                                 false, false);
}

// ---------------------------------------------------------------------------
__global__ void cvt_f32_bf16(const float* __restrict__ src, __bf16* __restrict__ dst,
                             int n) {
  int i = blockIdx.x * blockDim.x + threadIdx.x;
  int stride = gridDim.x * blockDim.x;
  for (; i < n; i += stride) dst[i] = (__bf16)src[i];
}

// ---------------------------------------------------------------------------
// Out[M,N] = A[M,Kd] * W[N,Kd]^T + bias.  Block: 256 thr (8 waves) -> 64x128
// tile; wave -> 32x32 (2x2 WMMA tiles).  Register-double-buffered LDS staging:
// next K-slice global loads overlap current slice's WMMAs.
template <bool OUT_BF16>
__global__ __launch_bounds__(256) void gemm_bf16(const __bf16* __restrict__ A,
                                                 const __bf16* __restrict__ W,
                                                 const float* __restrict__ bias,
                                                 void* __restrict__ Out, int M, int N,
                                                 int Kd) {
  __shared__ alignas(16) __bf16 As[64 * 32];
  __shared__ alignas(16) __bf16 Bs[128 * 32];
  const int mb = blockIdx.y * 64;
  const int nb = blockIdx.x * 128;
  const int t = threadIdx.x;
  const int wi = t >> 5, lane = t & 31, l16 = lane & 15;
  const int kb8 = (lane < 16) ? 0 : 8;
  const int kb16 = (lane < 16) ? 0 : 16;
  const int rowoff = (lane < 16) ? 0 : 8;
  const int rm = (wi >> 2) * 32;  // wave row offset (0 or 32)
  const int cn = (wi & 3) * 32;   // wave col offset (0..96)
  const int arow = t >> 2, acg = (t & 3) * 8;   // A staging: 64 rows x 32
  const int brow = t >> 1, bcg = (t & 1) * 16;  // B staging: 128 rows x 32

  const __bf16* ap = A + (size_t)(mb + arow) * Kd + acg;
  const __bf16* bp = W + (size_t)(nb + brow) * Kd + bcg;

  v8f acc[2][2] = {};
  bf8v a_reg = *(const bf8v*)(ap);
  bf8v b_reg0 = *(const bf8v*)(bp);
  bf8v b_reg1 = *(const bf8v*)(bp + 8);

  for (int kc = 0; kc < Kd; kc += 32) {
    *(bf8v*)(As + arow * 32 + acg) = a_reg;
    *(bf8v*)(Bs + brow * 32 + bcg) = b_reg0;
    *(bf8v*)(Bs + brow * 32 + bcg + 8) = b_reg1;
    __syncthreads();

    const int kn = kc + 32;
    if (kn < Kd) {  // prefetch next K slice (overlaps with WMMAs below)
      a_reg = *(const bf8v*)(ap + kn);
      b_reg0 = *(const bf8v*)(bp + kn);
      b_reg1 = *(const bf8v*)(bp + kn + 8);
    }

    v16bf a0 = load_frag_a(As + rm * 32, 32, l16, kb8);
    v16bf a1 = load_frag_a(As + (rm + 16) * 32, 32, l16, kb8);
    v16bf b0 = load_frag_b(Bs + cn * 32, 32, l16, kb16);
    v16bf b1 = load_frag_b(Bs + (cn + 16) * 32, 32, l16, kb16);
    acc[0][0] = wmma_bf16(a0, b0, acc[0][0]);
    acc[0][1] = wmma_bf16(a0, b1, acc[0][1]);
    acc[1][0] = wmma_bf16(a1, b0, acc[1][0]);
    acc[1][1] = wmma_bf16(a1, b1, acc[1][1]);
    __syncthreads();
  }

#pragma unroll
  for (int tm = 0; tm < 2; ++tm)
#pragma unroll
    for (int tn = 0; tn < 2; ++tn)
#pragma unroll
      for (int r = 0; r < 8; ++r) {
        int gm = mb + rm + tm * 16 + r + rowoff;
        int gn = nb + cn + tn * 16 + l16;
        float v = acc[tm][tn][r] + bias[gn];
        if constexpr (OUT_BF16)
          ((__bf16*)Out)[(size_t)gm * N + gn] = (__bf16)v;
        else
          ((float*)Out)[(size_t)gm * N + gn] = v;
      }
}

// ---------------------------------------------------------------------------
// RoPE (faithful: theta = 10000^(-(2^p)/d), 1-indexed positions) + head split.
// Writes Q,K roped as [B,H,T,HD] bf16 and V transposed as [B,H,HD,T] bf16.
__global__ __launch_bounds__(256) void rope_kernel(const __bf16* __restrict__ qkv,
                                                   __bf16* __restrict__ Qr,
                                                   __bf16* __restrict__ Kr,
                                                   __bf16* __restrict__ Vt) {
  int idx = blockIdx.x * 256 + threadIdx.x;  // B*H*T*64 threads
  int p = idx & 63;
  int t = (idx >> 6) & (Tz - 1);
  int h = (idx >> 17) & (Hz - 1);
  int b = idx >> 21;

  float theta = powf(10000.0f, -exp2f((float)p) * (1.0f / (float)HDz));
  float ang = (float)(t + 1) * theta;
  float cs = cosf(ang), sn = sinf(ang);

  size_t row = (size_t)(b * Tz + t) * C3z + h * HDz;
  float q0 = (float)qkv[row + p];
  float q1 = (float)qkv[row + p + 64];
  float k0 = (float)qkv[row + Cz + p];
  float k1 = (float)qkv[row + Cz + p + 64];
  float v0 = (float)qkv[row + 2 * Cz + p];
  float v1 = (float)qkv[row + 2 * Cz + p + 64];

  size_t bh = (size_t)(b * Hz + h);
  size_t qk = (bh * Tz + t) * HDz;
  Qr[qk + p] = (__bf16)(q0 * cs - q1 * sn);
  Qr[qk + p + 64] = (__bf16)(q1 * cs + q0 * sn);
  Kr[qk + p] = (__bf16)(k0 * cs - k1 * sn);
  Kr[qk + p + 64] = (__bf16)(k1 * cs + k0 * sn);
  Vt[(bh * HDz + p) * Tz + t] = (__bf16)v0;
  Vt[(bh * HDz + p + 64) * Tz + t] = (__bf16)v1;
}

// ---------------------------------------------------------------------------
// Flash attention: one wave per 16-row query tile of one (b,h). Streams keys
// in blocks of 32 with online softmax.  Q: [B,H,T,HD], K: [B,H,T,HD],
// Vt: [B,H,HD,T] (all bf16).  Y out: [B*T, C] bf16.
__global__ __launch_bounds__(256) void attn_kernel(const __bf16* __restrict__ Q,
                                                   const __bf16* __restrict__ K,
                                                   const __bf16* __restrict__ Vt,
                                                   __bf16* __restrict__ Y) {
  __shared__ alignas(16) __bf16 Pl[8][16 * 32];  // per-wave P transpose buffer
  const int wi = threadIdx.x >> 5, lane = threadIdx.x & 31, l16 = lane & 15;
  const int kb8 = (lane < 16) ? 0 : 8;
  const int kb16 = (lane < 16) ? 0 : 16;
  const int rowoff = (lane < 16) ? 0 : 8;

  int wid = blockIdx.x * 8 + wi;        // 4096 waves total
  const int qt = wid & (Tz / 16 - 1);   // 128 query tiles
  const int h = (wid >> 7) & (Hz - 1);
  const int b = wid >> 11;
  const int qbase = qt * 16;

  const __bf16* Qbh = Q + (size_t)(b * Hz + h) * Tz * HDz;
  const __bf16* Kbh = K + (size_t)(b * Hz + h) * Tz * HDz;
  const __bf16* Vbh = Vt + (size_t)(b * Hz + h) * HDz * Tz;
  __bf16* pl = &Pl[wi][0];

  v16bf qf[4];
#pragma unroll
  for (int c = 0; c < 4; ++c)
    qf[c] = load_frag_a(Qbh + (size_t)qbase * HDz + c * 32, HDz, l16, kb8);

  v8f o[8] = {};
  float mrun[8], lrun[8];
#pragma unroll
  for (int r = 0; r < 8; ++r) { mrun[r] = -1e30f; lrun[r] = 0.0f; }

  const float scale = 0.08838834764831845f;  // 1/sqrt(128)

  for (int jb = 0; jb <= qbase + 15; jb += 32) {
    // --- preload all K fragments (one load clause, no per-WMMA stalls) ---
    v16bf kf0[4], kf1[4];
#pragma unroll
    for (int c = 0; c < 4; ++c) {
      kf0[c] = load_frag_b(Kbh + (size_t)jb * HDz + c * 32, HDz, l16, kb16);
      kf1[c] = load_frag_b(Kbh + (size_t)(jb + 16) * HDz + c * 32, HDz, l16, kb16);
    }
    // --- S = Q * K^T for 32 keys (two 16-col tiles), contraction over HD ---
    v8f s0 = {}, s1 = {};
#pragma unroll
    for (int c = 0; c < 4; ++c) {
      s0 = wmma_bf16(qf[c], kf0[c], s0);
      s1 = wmma_bf16(qf[c], kf1[c], s1);
    }
    // --- prefetch V fragments; loads fly while softmax VALU work runs ---
    v16bf vf[8];
#pragma unroll
    for (int t8 = 0; t8 < 8; ++t8)
      vf[t8] = load_frag_b(Vbh + (size_t)t8 * 16 * Tz + jb, Tz, l16, kb16);

    // --- causal mask + online softmax (rows live on 16-lane halves) ---
    float p0[8], p1[8], al[8];
#pragma unroll
    for (int r = 0; r < 8; ++r) {
      int qrow = qbase + r + rowoff;
      int key0 = jb + l16, key1 = key0 + 16;
      float x0 = (key0 <= qrow) ? s0[r] * scale : -1e30f;
      float x1 = (key1 <= qrow) ? s1[r] * scale : -1e30f;
      float mx = fmaxf(x0, x1);
      mx = fmaxf(mx, __shfl_xor(mx, 1, 32));
      mx = fmaxf(mx, __shfl_xor(mx, 2, 32));
      mx = fmaxf(mx, __shfl_xor(mx, 4, 32));
      mx = fmaxf(mx, __shfl_xor(mx, 8, 32));
      float mnew = fmaxf(mrun[r], mx);
      float a = __expf(mrun[r] - mnew);
      float e0 = __expf(x0 - mnew);
      float e1 = __expf(x1 - mnew);
      float rs = e0 + e1;
      rs += __shfl_xor(rs, 1, 32);
      rs += __shfl_xor(rs, 2, 32);
      rs += __shfl_xor(rs, 4, 32);
      rs += __shfl_xor(rs, 8, 32);
      lrun[r] = lrun[r] * a + rs;
      mrun[r] = mnew;
      al[r] = a; p0[r] = e0; p1[r] = e1;
    }
#pragma unroll
    for (int t8 = 0; t8 < 8; ++t8)
#pragma unroll
      for (int r = 0; r < 8; ++r) o[t8][r] *= al[r];

    // --- P (C-layout) -> LDS -> A-layout fragment ---
#pragma unroll
    for (int r = 0; r < 8; ++r) {
      int m = r + rowoff;
      pl[m * 32 + l16] = (__bf16)p0[r];
      pl[m * 32 + 16 + l16] = (__bf16)p1[r];
    }
    asm volatile("s_wait_dscnt 0" ::: "memory");
    v16bf pf = load_frag_a(pl, 32, l16, kb8);

    // --- O += P * V ; V fragments already resident in VGPRs ---
#pragma unroll
    for (int t8 = 0; t8 < 8; ++t8) o[t8] = wmma_bf16(pf, vf[t8], o[t8]);
  }

  float linv[8];
#pragma unroll
  for (int r = 0; r < 8; ++r) linv[r] = 1.0f / lrun[r];
#pragma unroll
  for (int t8 = 0; t8 < 8; ++t8)
#pragma unroll
    for (int r = 0; r < 8; ++r) {
      size_t row = (size_t)(b * Tz + qbase + r + rowoff);
      Y[row * Cz + h * HDz + t8 * 16 + l16] = (__bf16)(o[t8][r] * linv[r]);
    }
}

// ---------------------------------------------------------------------------
extern "C" void kernel_launch(void* const* d_in, const int* in_sizes, int n_in,
                              void* d_out, int out_size, void* d_ws, size_t ws_size,
                              hipStream_t stream) {
  const float* x = (const float*)d_in[0];     // [B,T,C]
  const float* Wqkv = (const float*)d_in[1];  // [3C,C]
  const float* bqkv = (const float*)d_in[2];  // [3C]
  const float* Wo = (const float*)d_in[3];    // [C,C]
  const float* bo = (const float*)d_in[4];    // [C]
  float* out = (float*)d_out;                 // [B,T,C]

  char* ws = (char*)d_ws;
  const size_t M = (size_t)Bz * Tz;  // 4096
  size_t o_xbf = 0;                                  // X bf16 [M,C] (reused as Y)
  size_t o_wqkv = o_xbf + M * Cz * 2;                // Wqkv bf16 [3C,C]
  size_t o_wo = o_wqkv + (size_t)C3z * Cz * 2;       // Wo bf16 [C,C]
  size_t o_qkv = o_wo + (size_t)Cz * Cz * 2;         // QKV bf16 [M,3C]
  size_t o_q = o_qkv + M * C3z * 2;                  // Q roped [B,H,T,HD]
  size_t o_k = o_q + M * Cz * 2;                     // K roped
  size_t o_v = o_k + M * Cz * 2;                     // V^T [B,H,HD,T]

  __bf16* Xbf = (__bf16*)(ws + o_xbf);
  __bf16* Wqkvbf = (__bf16*)(ws + o_wqkv);
  __bf16* Wobf = (__bf16*)(ws + o_wo);
  __bf16* QKVbf = (__bf16*)(ws + o_qkv);
  __bf16* Qr = (__bf16*)(ws + o_q);
  __bf16* Kr = (__bf16*)(ws + o_k);
  __bf16* Vt = (__bf16*)(ws + o_v);
  __bf16* Ybf = Xbf;  // alias: X is dead after QKV GEMM

  // 1) f32 -> bf16 conversions
  cvt_f32_bf16<<<2048, 256, 0, stream>>>(x, Xbf, (int)(M * Cz));
  cvt_f32_bf16<<<2048, 256, 0, stream>>>(Wqkv, Wqkvbf, C3z * Cz);
  cvt_f32_bf16<<<2048, 256, 0, stream>>>(Wo, Wobf, Cz * Cz);

  // 2) QKV projection: [M,3C] = X @ Wqkv^T + bqkv
  gemm_bf16<true><<<dim3(C3z / 128, M / 64), 256, 0, stream>>>(
      Xbf, Wqkvbf, bqkv, QKVbf, (int)M, C3z, Cz);

  // 3) RoPE + head reorg (V transposed)
  rope_kernel<<<(Bz * Hz * Tz * 64) / 256, 256, 0, stream>>>(QKVbf, Qr, Kr, Vt);

  // 4) Flash attention -> Y [M,C] bf16
  attn_kernel<<<(Bz * Hz * (Tz / 16)) / 8, 256, 0, stream>>>(Qr, Kr, Vt, Ybf);

  // 5) Output projection: out = Y @ Wo^T + bo (f32)
  gemm_bf16<false><<<dim3(Cz / 128, M / 64), 256, 0, stream>>>(
      Ybf, Wobf, bo, out, (int)M, Cz, Cz);
}
